// SpUNetBase_43636867728112
// MI455X (gfx1250) — compile-verified, matrix operand order checked by reading, not run on hardware
//
#include <hip/hip_runtime.h>
#include <hip/hip_bf16.h>

// ---------------------------------------------------------------------------
// Sparse 3D U-Net (SpUNet) forward for MI455X / gfx1250.
// Weights pre-packed to f16 (transposed+padded) once per layer; conv kernel:
// gather (b128) -> LDS-staged WMMA (f16 in / f32 acc) -> atomic scatter.
// ---------------------------------------------------------------------------

typedef __attribute__((ext_vector_type(16))) _Float16 v16h;
typedef __attribute__((ext_vector_type(8)))  _Float16 v8h;
typedef __attribute__((ext_vector_type(4)))  _Float16 v4h;
typedef __attribute__((ext_vector_type(8)))  float    v8f;

#define THREADS   256
#define WAVES     8            // 256 / wave32

// ---------------------------------------------------------------------------
// One-shot weight pack: W (nk x Ci x Co, f32) -> Wt (nk x CoPad x Kpad, f16),
// transposed (col-major per offset) and zero-padded.
// ---------------------------------------------------------------------------
__global__ void pack_w(const float* __restrict__ W, int nk, int Ci, int Co,
                       int Kpad, int CoPad, _Float16* __restrict__ Wt)
{
    const long long total = (long long)nk * CoPad * Kpad;
    const long long i = (long long)blockIdx.x * blockDim.x + threadIdx.x;
    if (i < total) {
        const int k = (int)(i % Kpad);
        const long long t = i / Kpad;
        const int c  = (int)(t % CoPad);
        const int ko = (int)(t / CoPad);
        const float v = (k < Ci && c < Co)
                      ? W[((size_t)ko * Ci + k) * Co + c] : 0.0f;
        Wt[i] = (_Float16)v;
    }
}

// ---------------------------------------------------------------------------
// Tiled sparse conv (one kernel offset):  out[iout[r],:] += feat[iin[r],:] @ W
// 256 threads = 8 waves; wave w owns rows [blockIdx.x*128 + w*16, +16).
//   Bsh : pre-packed f16 weight tile, copied b128 global -> b128 LDS
//   Ash : gathered feature rows (float4 -> 4xf16), zero-padded
// Compute: per Co-tile, per 32-K chunk: 4x ds_load_b128 -> v_wmma_f32_16x16x32_f16
// (fully unrolled).  dense==1: identity rows, direct store (proj / final 1x1).
// ---------------------------------------------------------------------------
template <int KPAD, int COPAD>
__global__ __launch_bounds__(THREADS)
void sconv_wmma(const float* __restrict__ feat, const _Float16* __restrict__ Wt,
                const int* __restrict__ iin, const int* __restrict__ iout,
                int L, int Ci, int Co, float* __restrict__ out, int dense)
{
    __shared__ __align__(16) _Float16 Bsh[COPAD * KPAD];
    __shared__ __align__(16) _Float16 Ash[WAVES * 16 * KPAD];

    const int tid  = threadIdx.x;
    const int w    = tid >> 5;
    const int lane = tid & 31;
    const int rloc = lane & 15;
    const int hi   = lane >> 4;

    __builtin_prefetch(Wt, 0, 1);   // global_prefetch_b8

    // ---- stage packed W tile: straight b128 copy ----------------------------
    {
        constexpr int NB8 = (COPAD * KPAD) / 8;   // v8h chunks
        const v8h* src = (const v8h*)Wt;
        v8h* dst = (v8h*)Bsh;
#pragma unroll
        for (int idx = tid; idx < NB8; idx += THREADS)
            dst[idx] = src[idx];                  // global_load_b128 -> ds_store_b128
    }

    // ---- stage gathered A rows (per wave, b128 gather along channels) -------
    const int rowbase = blockIdx.x * (WAVES * 16) + w * 16;
    const bool vec4 = ((Ci & 3) == 0);
    const int nv = Ci >> 2;
#pragma unroll 4
    for (int row = 0; row < 16; ++row) {
        const int r = rowbase + row;              // wave-uniform
        _Float16* dst = &Ash[(w * 16 + row) * KPAD];
        if (r < L) {
            const int src = dense ? r : iin[r];
            if (vec4) {
                const float4* srow = (const float4*)(feat + (size_t)src * Ci);
#pragma unroll
                for (int q = lane; q < (KPAD >> 2); q += 32) {
                    float4 f = {0.0f, 0.0f, 0.0f, 0.0f};
                    if (q < nv) f = srow[q];      // global_load_b128
                    v4h h;
                    h[0] = (_Float16)f.x; h[1] = (_Float16)f.y;
                    h[2] = (_Float16)f.z; h[3] = (_Float16)f.w;
                    *(v4h*)(dst + 4 * q) = h;     // ds_store_b64
                }
            } else {                              // stem (Ci==6) fallback
                const float* srow = feat + (size_t)src * Ci;
#pragma unroll
                for (int k = lane; k < KPAD; k += 32)
                    dst[k] = (_Float16)((k < Ci) ? srow[k] : 0.0f);
            }
        } else {
#pragma unroll
            for (int q = lane; q < (KPAD >> 2); q += 32) {
                v4h h = {};
                *(v4h*)(dst + 4 * q) = h;
            }
        }
    }
    __syncthreads();

    // ---- WMMA compute: fully unrolled (KPAD/32) x (COPAD/16) nest -----------
    const _Float16* arow = &Ash[(w * 16 + rloc) * KPAD];
#pragma unroll
    for (int t = 0; t < (COPAD >> 4); ++t) {
        const _Float16* bcol = &Bsh[(t * 16 + rloc) * KPAD];
        v8f acc = {};
#pragma unroll
        for (int k0 = 0; k0 < KPAD; k0 += 32) {
            const int kb = k0 + hi * 8;
            const v8h alo = *(const v8h*)(arow + kb);
            const v8h ahi = *(const v8h*)(arow + kb + 16);
            const v8h blo = *(const v8h*)(bcol + kb);
            const v8h bhi = *(const v8h*)(bcol + kb + 16);
            v16h a, b;
#pragma unroll
            for (int j = 0; j < 8; ++j) {
                a[j] = alo[j]; a[j + 8] = ahi[j];
                b[j] = blo[j]; b[j + 8] = bhi[j];
            }
            acc = __builtin_amdgcn_wmma_f32_16x16x32_f16(false, a, false, b,
                                                         (short)0, acc, false, false);
        }
        const int col = t * 16 + rloc;
        if (col < Co) {
            if (dense) {
#pragma unroll
                for (int v = 0; v < 8; ++v) {
                    const int rr = rowbase + v + hi * 8;
                    if (rr < L) out[(size_t)rr * Co + col] = acc[v];
                }
            } else {
#pragma unroll
                for (int v = 0; v < 8; ++v) {
                    const int rr = rowbase + v + hi * 8;
                    if (rr < L) atomicAdd(&out[(size_t)iout[rr] * Co + col], acc[v]);
                }
            }
        }
    }
}

// BN statistics: per-channel sum / sumsq via LDS ds_add_f32 + global atomics.
__global__ void bn_stats(const float* __restrict__ x, int n, int C,
                         float* __restrict__ gsum, float* __restrict__ gsq)
{
    __shared__ float ls[128];
    __shared__ float lq[128];
    for (int c = threadIdx.x; c < 128; c += blockDim.x) { ls[c] = 0.f; lq[c] = 0.f; }
    __syncthreads();

    const int r = blockIdx.x * blockDim.x + threadIdx.x;
    if (r < n) {
        const float* row = x + (size_t)r * C;
        for (int c = 0; c < C; ++c) {
            const float v = row[c];
            atomicAdd(&ls[c], v);
            atomicAdd(&lq[c], v * v);
        }
    }
    __syncthreads();
    for (int c = threadIdx.x; c < C; c += blockDim.x) {
        atomicAdd(&gsum[c], ls[c]);
        atomicAdd(&gsq[c],  lq[c]);
    }
}

__global__ void bn_finalize(const float* __restrict__ gsum, const float* __restrict__ gsq,
                            const float* __restrict__ g, const float* __restrict__ b,
                            int n, int C, float* __restrict__ scale, float* __restrict__ shift)
{
    const int c = threadIdx.x;
    if (c < C) {
        const float inv = 1.0f / (float)n;
        const float mu  = gsum[c] * inv;
        const float var = gsq[c] * inv - mu * mu;
        const float s   = g[c] * rsqrtf(var + 1e-3f);
        scale[c] = s;
        shift[c] = b[c] - mu * s;
    }
}

__global__ void bn_apply(float* __restrict__ x, long long total, int C,
                         const float* __restrict__ scale, const float* __restrict__ shift,
                         int relu)
{
    const long long i = (long long)blockIdx.x * blockDim.x + threadIdx.x;
    if (i < total) {
        const int c = (int)(i % C);
        float v = x[i] * scale[c] + shift[c];
        if (relu) v = fmaxf(v, 0.0f);
        x[i] = v;
    }
}

__global__ void add_relu_k(float* __restrict__ a, const float* __restrict__ b, long long n)
{
    const long long i = (long long)blockIdx.x * blockDim.x + threadIdx.x;
    if (i < n) a[i] = fmaxf(a[i] + b[i], 0.0f);
}

__global__ void concat2(const float* __restrict__ a, int Ca,
                        const float* __restrict__ b, int Cb,
                        float* __restrict__ out, int n)
{
    const int C = Ca + Cb;
    const long long total = (long long)n * C;
    const long long i = (long long)blockIdx.x * blockDim.x + threadIdx.x;
    if (i < total) {
        const int r = (int)(i / C);
        const int c = (int)(i % C);
        out[i] = (c < Ca) ? a[(size_t)r * Ca + c] : b[(size_t)r * Cb + (c - Ca)];
    }
}

// ---------------------------------------------------------------------------
// Host orchestration
// ---------------------------------------------------------------------------
struct KMap { const int* in[27]; const int* out[27]; int len[27]; int nk; };
struct Blk  { const float *c1, *g1, *b1, *c2, *g2, *b2, *pw, *pg, *pb; };

static void launch_conv(hipStream_t stream, const float* x, const _Float16* Wt,
                        const int* iin, const int* iout, int L, int Ci, int Co,
                        float* y, int dense)
{
    const dim3 g((unsigned)((L + WAVES * 16 - 1) / (WAVES * 16)), 1, 1);
    const dim3 b(THREADS, 1, 1);
    const int Kpad  = (Ci + 31) & ~31;
    const int CoPad = (Co + 15) & ~15;
#define SCONV_CASE(KP, CP)                                                      \
    if (Kpad == KP && CoPad == CP) {                                            \
        sconv_wmma<KP, CP><<<g, b, 0, stream>>>(x, Wt, iin, iout, L, Ci, Co, y, \
                                                dense);                         \
        return;                                                                 \
    }
    SCONV_CASE(32, 32) SCONV_CASE(32, 48) SCONV_CASE(32, 64)
    SCONV_CASE(64, 32) SCONV_CASE(64, 48) SCONV_CASE(64, 64)
    SCONV_CASE(96, 32) SCONV_CASE(96, 48) SCONV_CASE(96, 64)
#undef SCONV_CASE
}

extern "C" void kernel_launch(void* const* d_in, const int* in_sizes, int n_in,
                              void* d_out, int out_size, void* d_ws, size_t ws_size,
                              hipStream_t stream)
{
    (void)out_size;
    if (n_in < 267) return;   // flattened pytree: feat, 70 params, 194 kmap arrays, n1, n2

    const float* feat = (const float*)d_in[0];
    const int n0 = in_sizes[0] / 6;
    const int n1 = in_sizes[141 + 13 * 2];   // subm1 center-offset pair length == n1
    const int n2 = in_sizes[211 + 13 * 2];   // subm2 center-offset pair length == n2
    if (n0 <= 0 || n1 <= 0 || n2 <= 0) return;

    auto F = [&](int i) -> const float* { return (const float*)d_in[i]; };

    auto mkmap = [&](int base, int nk, bool inverse) -> KMap {
        KMap m; m.nk = nk;
        for (int k = 0; k < nk; ++k) {
            const int* a = (const int*)d_in[base + 2 * k];
            const int* b = (const int*)d_in[base + 2 * k + 1];
            m.in[k]  = inverse ? b : a;
            m.out[k] = inverse ? a : b;
            m.len[k] = in_sizes[base + 2 * k];
        }
        return m;
    };
    const KMap subm0 = mkmap(71, 27, false);
    const KMap dmap0 = mkmap(125, 8, false);
    const KMap inv0  = mkmap(125, 8, true);
    const KMap subm1 = mkmap(141, 27, false);
    const KMap dmap1 = mkmap(195, 8, false);
    const KMap inv1  = mkmap(195, 8, true);
    const KMap subm2 = mkmap(211, 27, false);

    // ---- workspace carve-out (16B-aligned slabs) -----------------------------
    float* ws = (float*)d_ws;
    size_t off = 0;
    auto alloc = [&](size_t nElem) -> float* {
        off = (off + 3) & ~(size_t)3;
        float* p = ws + off; off += nElem; return p;
    };
    float*     skip0 = alloc((size_t)n0 * 32);
    float*     skip1 = alloc((size_t)n1 * 32);
    float*     bufA  = alloc((size_t)n0 * 96);
    float*     bufB  = alloc((size_t)n0 * 96);
    float*     bufC  = alloc((size_t)n0 * 96);
    float*     stats = alloc(256);                 // sum[128] | sumsq[128]
    float*     bnsc  = alloc(256);                 // scale[128] | shift[128]
    _Float16*  wtbuf = (_Float16*)alloc(27 * 96 * 64 / 2 + 16);  // packed weights
    if (off * sizeof(float) > ws_size) return;

    // ---- stream helpers ------------------------------------------------------
    auto zero = [&](float* p, size_t nel) {
        hipMemsetAsync(p, 0, nel * sizeof(float), stream);
    };
    auto pack = [&](const float* W, int nk, int Ci, int Co) {
        const int Kpad  = (Ci + 31) & ~31;
        const int CoPad = (Co + 15) & ~15;
        const long long tot = (long long)nk * CoPad * Kpad;
        pack_w<<<(unsigned)((tot + THREADS - 1) / THREADS), THREADS, 0, stream>>>(
            W, nk, Ci, Co, Kpad, CoPad, wtbuf);
        return (size_t)CoPad * Kpad;               // per-offset stride in halves
    };
    auto sconv = [&](const float* x, const float* W, const KMap& m,
                     int Ci, int Co, float* y, int nOut) {
        const size_t stride = pack(W, m.nk, Ci, Co);
        zero(y, (size_t)nOut * Co);
        for (int k = 0; k < m.nk; ++k) {
            const int L = m.len[k];
            if (L <= 0) continue;
            launch_conv(stream, x, wtbuf + (size_t)k * stride, m.in[k], m.out[k],
                        L, Ci, Co, y, 0);
        }
    };
    auto gemm = [&](const float* x, const float* W, int N, int Ci, int Co, float* y) {
        pack(W, 1, Ci, Co);
        launch_conv(stream, x, wtbuf, nullptr, nullptr, N, Ci, Co, y, 1);
    };
    auto bn = [&](float* x, int N, int C, const float* g_, const float* b_, bool relu) {
        hipMemsetAsync(stats, 0, 256 * sizeof(float), stream);
        bn_stats<<<(unsigned)((N + THREADS - 1) / THREADS), THREADS, 0, stream>>>(
            x, N, C, stats, stats + 128);
        bn_finalize<<<1, 128, 0, stream>>>(stats, stats + 128, g_, b_, N, C,
                                           bnsc, bnsc + 128);
        const long long tot = (long long)N * C;
        bn_apply<<<(unsigned)((tot + THREADS - 1) / THREADS), THREADS, 0, stream>>>(
            x, tot, C, bnsc, bnsc + 128, relu ? 1 : 0);
    };
    auto addrelu = [&](float* a, const float* b, long long nel) {
        add_relu_k<<<(unsigned)((nel + THREADS - 1) / THREADS), THREADS, 0, stream>>>(a, b, nel);
    };
    auto concat = [&](const float* a, int Ca, const float* b, int Cb, float* y, int n) {
        const long long tot = (long long)n * (Ca + Cb);
        concat2<<<(unsigned)((tot + THREADS - 1) / THREADS), THREADS, 0, stream>>>(
            a, Ca, b, Cb, y, n);
    };
    auto mkblk = [&](int base, bool proj) -> Blk {
        Blk o;
        o.c1 = F(base);     o.g1 = F(base + 1); o.b1 = F(base + 2);
        o.c2 = F(base + 3); o.g2 = F(base + 4); o.b2 = F(base + 5);
        o.pw = proj ? F(base + 6) : nullptr;
        o.pg = proj ? F(base + 7) : nullptr;
        o.pb = proj ? F(base + 8) : nullptr;
        return o;
    };
    auto doBlock = [&](float*& x, float*& t1, float*& t2, int N, int Ci, int Co,
                       const Blk& P, const KMap& m) {
        sconv(x, P.c1, m, Ci, Co, t1, N);  bn(t1, N, Co, P.g1, P.b1, true);
        sconv(t1, P.c2, m, Co, Co, t2, N); bn(t2, N, Co, P.g2, P.b2, false);
        if (P.pw) {
            gemm(x, P.pw, N, Ci, Co, t1);
            bn(t1, N, Co, P.pg, P.pb, false);
            addrelu(t2, t1, (long long)N * Co);
        } else {
            addrelu(t2, x, (long long)N * Co);
        }
        float* o = x; x = t2; t2 = o;   // result becomes new x
    };

    // ---- forward pass --------------------------------------------------------
    float* x  = bufA;
    float* t1 = bufB;
    float* t2 = bufC;

    // stem (6 -> 32) at n0, BN+ReLU; keep as skip0
    sconv(feat, F(1), subm0, 6, 32, skip0, n0);
    bn(skip0, n0, 32, F(2), F(3), true);

    // down0 (32 -> 32) to n1
    sconv(skip0, F(4), dmap0, 32, 32, x, n1);
    bn(x, n1, 32, F(5), F(6), true);

    // enc0: two residual blocks 32->32 on subm1
    doBlock(x, t1, t2, n1, 32, 32, mkblk(7, false),  subm1);
    doBlock(x, t1, t2, n1, 32, 32, mkblk(13, false), subm1);

    // skip1 snapshot
    hipMemcpyAsync(skip1, x, (size_t)n1 * 32 * sizeof(float),
                   hipMemcpyDeviceToDevice, stream);

    // down1 (32 -> 64) to n2
    sconv(x, F(19), dmap1, 32, 64, t1, n2);
    bn(t1, n2, 64, F(20), F(21), true);
    { float* o = x; x = t1; t1 = o; }

    // enc1: two residual blocks 64->64 on subm2
    doBlock(x, t1, t2, n2, 64, 64, mkblk(22, false), subm2);
    doBlock(x, t1, t2, n2, 64, 64, mkblk(28, false), subm2);

    // up1: inverse conv (64 -> 64) coarse->fine to n1
    sconv(x, F(34), inv1, 64, 64, t1, n1);
    bn(t1, n1, 64, F(35), F(36), true);
    { float* o = x; x = t1; t1 = o; }

    // concat [x(n1x64), skip1(n1x32)] -> n1x96
    concat(x, 64, skip1, 32, t1, n1);
    { float* o = x; x = t1; t1 = o; }

    // dec1: block(96->64 w/ proj), block(64->64)
    doBlock(x, t1, t2, n1, 96, 64, mkblk(37, true),  subm1);
    doBlock(x, t1, t2, n1, 64, 64, mkblk(46, false), subm1);

    // up0: inverse conv (64 -> 48) to n0
    sconv(x, F(52), inv0, 64, 48, t1, n0);
    bn(t1, n0, 48, F(53), F(54), true);
    { float* o = x; x = t1; t1 = o; }

    // concat [x(n0x48), skip0(n0x32)] -> n0x80
    concat(x, 48, skip0, 32, t1, n0);
    { float* o = x; x = t1; t1 = o; }

    // dec0: block(80->48 w/ proj), block(48->48)
    doBlock(x, t1, t2, n0, 80, 48, mkblk(55, true),  subm0);
    doBlock(x, t1, t2, n0, 48, 48, mkblk(64, false), subm0);

    // final 1x1: (n0 x 48) @ (48 x 20) -> d_out
    gemm(x, F(70), n0, 48, 20, (float*)d_out);
}
